// GCNEncoder_12799002542721
// MI455X (gfx1250) — compile-verified, matrix operand order checked by reading, not run on hardware
//
#include <hip/hip_runtime.h>
#include <hip/hip_bf16.h>
#include <math.h>
#include <stdint.h>

typedef __attribute__((ext_vector_type(16))) _Float16 v16h;
typedef __attribute__((ext_vector_type(8)))  _Float16 v8h;
typedef __attribute__((ext_vector_type(8)))  float    v8f;

#define D_IN   256
#define QKV_LD 768
#define NHEAD  4
#define HDIM   64
#define PART   4      // key-dimension partitions for flash attention

#if defined(__has_builtin)
#if __has_builtin(__builtin_amdgcn_tensor_load_to_lds)
#define HAVE_TDM 1
#endif
#endif
#ifndef HAVE_TDM
#define HAVE_TDM 0
#endif

// ---------------------------------------------------------------------------
// TDM: DMA a 2D f32 tile (rows x rowFloats, row stride strideFloats) from
// global memory into LDS.  D# built per CDNA5 ISA §8.3/§8.4.  Wave-level op:
// call from one wave with EXEC all-ones; follow with s_wait_tensorcnt 0.
// ---------------------------------------------------------------------------
__device__ __forceinline__ void tdm_load_tile_f32(float* ldsDst, const float* gsrc,
                                                  uint32_t rowFloats, uint32_t rows,
                                                  uint32_t strideFloats)
{
#if HAVE_TDM
    typedef uint32_t v4u __attribute__((ext_vector_type(4)));
    typedef int      v8i __attribute__((ext_vector_type(8)));
    typedef int      v4i __attribute__((ext_vector_type(4)));

    const uint64_t ga  = (uint64_t)(uintptr_t)gsrc;
    const uint32_t lds = (uint32_t)(uintptr_t)ldsDst;   // low 32 bits = LDS offset

    v4u g0;
    g0[0] = 1u;                                         // count=1 (valid), user mode
    g0[1] = lds;                                        // lds_addr (bytes)
    g0[2] = (uint32_t)ga;                               // global_addr[31:0]
    g0[3] = (uint32_t)((ga >> 32) & 0x01FFFFFFu) | (2u << 30);  // addr[56:32] | type=2

    v8i g1;
    g1[0] = (int)(2u << 16);                            // wg_mask=0, data_size=4B
    g1[1] = (int)((rowFloats & 0xFFFFu) << 16);         // barrier_addr=0 | td0[15:0]
    g1[2] = (int)(((rowFloats >> 16) & 0xFFFFu) |
                  ((rows & 0xFFFFu) << 16));            // td0[31:16] | td1[15:0]
    g1[3] = (int)(((rows >> 16) & 0xFFFFu) |
                  ((rowFloats & 0xFFFFu) << 16));       // td1[31:16] | tile_dim0
    g1[4] = (int)(rows & 0xFFFFu);                      // tile_dim1 | tile_dim2=0
    g1[5] = (int)strideFloats;                          // tensor_dim0_stride[31:0]
    g1[6] = 0;                                          // stride0[47:32] | stride1[15:0]
    g1[7] = 0;                                          // stride1[47:16]

    v4i g2 = {0, 0, 0, 0};
    v4i g3 = {0, 0, 0, 0};
#if __clang_major__ >= 23
    v8i g4 = {0, 0, 0, 0, 0, 0, 0, 0};
    __builtin_amdgcn_tensor_load_to_lds(g0, g1, g2, g3, g4, 0);
#else
    __builtin_amdgcn_tensor_load_to_lds(g0, g1, g2, g3, 0);
#endif
#endif
}

__device__ __forceinline__ void wait_tensorcnt0()
{
#if defined(__has_builtin) && __has_builtin(__builtin_amdgcn_s_wait_tensorcnt)
    __builtin_amdgcn_s_wait_tensorcnt(0);
#else
    asm volatile("s_wait_tensorcnt 0x0" ::: "memory");
#endif
}

// ---------------------------------------------------------------------------
// GEMM: C[M,Nc] = A[M,K] @ W[Nc,K]^T (+ bias).  One wave = 16x64 output tile:
// one A fragment is reused across four B fragments -> 4 v_wmma per K-step.
// ---------------------------------------------------------------------------
__global__ __launch_bounds__(32)
void gemm_wmma_f16(const float* __restrict__ A,
                   const float* __restrict__ W,
                   const float* __restrict__ bias,
                   float* __restrict__ C,
                   int M, int K, int Nc)
{
    const int tiles64 = Nc >> 6;
    const int tile    = blockIdx.x;
    const int tm      = (tile / tiles64) << 4;
    const int tn      = (tile % tiles64) << 6;
    const int lane    = threadIdx.x;
    const int half    = lane >> 4;
    const int l16     = lane & 15;
    const int kb      = half << 3;          // 0 or 8

    const float* __restrict__ arow = A + (size_t)(tm + l16) * K;
    const float* __restrict__ w0   = W + (size_t)(tn +  0 + l16) * K;
    const float* __restrict__ w1   = W + (size_t)(tn + 16 + l16) * K;
    const float* __restrict__ w2   = W + (size_t)(tn + 32 + l16) * K;
    const float* __restrict__ w3   = W + (size_t)(tn + 48 + l16) * K;

    v8f c0 = {}, c1 = {}, c2 = {}, c3 = {};
    for (int k0 = 0; k0 < K; k0 += 32) {
        v16h a, b0, b1, b2, b3;
#pragma unroll
        for (int e = 0; e < 8; ++e) {
            a[e]      = (_Float16)arow[k0 + kb + e];
            a[e + 8]  = (_Float16)arow[k0 + kb + 16 + e];
            b0[e]     = (_Float16)w0[k0 + kb + e];
            b0[e + 8] = (_Float16)w0[k0 + kb + 16 + e];
            b1[e]     = (_Float16)w1[k0 + kb + e];
            b1[e + 8] = (_Float16)w1[k0 + kb + 16 + e];
            b2[e]     = (_Float16)w2[k0 + kb + e];
            b2[e + 8] = (_Float16)w2[k0 + kb + 16 + e];
            b3[e]     = (_Float16)w3[k0 + kb + e];
            b3[e + 8] = (_Float16)w3[k0 + kb + 16 + e];
        }
        c0 = __builtin_amdgcn_wmma_f32_16x16x32_f16(false, a, false, b0, (short)0, c0, false, false);
        c1 = __builtin_amdgcn_wmma_f32_16x16x32_f16(false, a, false, b1, (short)0, c1, false, false);
        c2 = __builtin_amdgcn_wmma_f32_16x16x32_f16(false, a, false, b2, (short)0, c2, false, false);
        c3 = __builtin_amdgcn_wmma_f32_16x16x32_f16(false, a, false, b3, (short)0, c3, false, false);
    }

    const float bb0 = bias ? bias[tn +  0 + l16] : 0.0f;
    const float bb1 = bias ? bias[tn + 16 + l16] : 0.0f;
    const float bb2 = bias ? bias[tn + 32 + l16] : 0.0f;
    const float bb3 = bias ? bias[tn + 48 + l16] : 0.0f;
#pragma unroll
    for (int r = 0; r < 8; ++r) {
        float* __restrict__ crow = C + (size_t)(tm + r + 8 * half) * Nc + tn;
        crow[ 0 + l16] = c0[r] + bb0;
        crow[16 + l16] = c1[r] + bb1;
        crow[32 + l16] = c2[r] + bb2;
        crow[48 + l16] = c3[r] + bb3;
    }
}

// ---------------------------------------------------------------------------
// Partitioned flash attention.  Block = 128 threads (4 waves) = 64 query rows
// for one (head, key-partition).  K tile DMA'd into LDS by the Tensor Data
// Mover (f32); V tile cooperatively staged transposed as f16.  All fragment
// reads are aligned b128 LDS loads.
// ---------------------------------------------------------------------------
__global__ __launch_bounds__(128)
void attn_partial(const float* __restrict__ qkv,
                  float* __restrict__ opart,
                  float* __restrict__ mpart,
                  float* __restrict__ lpart,
                  int N)
{
    __shared__ __align__(16) float    kshF[32 * 64];       // [key][feat] f32 (TDM dest)
    __shared__ __align__(16) _Float16 vshT[64 * 32];       // [feat][key] f16
    __shared__ __align__(16) _Float16 pSh[4][16 * 32];     // per-wave P tile

    const int h     = blockIdx.y;
    const int p     = blockIdx.z;
    const int wid   = threadIdx.x >> 5;
    const int lane  = threadIdx.x & 31;
    const int half  = lane >> 4;
    const int l16   = lane & 15;
    const int kb    = half << 3;
    const int tid   = threadIdx.x;
    const int qrow0 = blockIdx.x * 64 + wid * 16;
    const float scale = 0.125f;                            // 1/sqrt(64)

    // ---- Q fragments: 16x64 as two 16x32 A-matrices (global, once) --------
    const float* __restrict__ qrow = qkv + (size_t)(qrow0 + l16) * QKV_LD + h * HDIM;
    v16h aq0, aq1;
#pragma unroll
    for (int e = 0; e < 8; ++e) {
        aq0[e]     = (_Float16)qrow[kb + e];
        aq0[e + 8] = (_Float16)qrow[kb + 16 + e];
        aq1[e]     = (_Float16)qrow[32 + kb + e];
        aq1[e + 8] = (_Float16)qrow[32 + kb + 16 + e];
    }

    v8f o0 = {}, o1 = {}, o2 = {}, o3 = {};
    float mrow[8], lrow[8];
#pragma unroll
    for (int r = 0; r < 8; ++r) { mrow[r] = -1e30f; lrow[r] = 0.0f; }

    const int j0 = p * (N / PART);
    const int j1 = j0 + (N / PART);

    // staging map: thread -> (key = tid/4, 16-feature segment = (tid&3)*16)
    const int skey = tid >> 2;
    const int sfs  = (tid & 3) << 4;

    for (int j = j0; j < j1; j += 32) {
        // ---- K tile: Tensor Data Mover (wave 0), 32 rows x 64 f32 ---------
#if HAVE_TDM
        if (wid == 0) {
            const float* ksrc = qkv + (size_t)j * QKV_LD + D_IN + h * HDIM;
            tdm_load_tile_f32(kshF, ksrc, HDIM, 32, QKV_LD);
            wait_tensorcnt0();
        }
#else
        {
            const float* __restrict__ krow = qkv + (size_t)(j + skey) * QKV_LD + D_IN + h * HDIM + sfs;
#pragma unroll
            for (int e = 0; e < 16; ++e)
                kshF[skey * 64 + sfs + e] = krow[e];
        }
#endif
        // ---- V tile: cooperative transpose-stage to f16 -------------------
        {
            const float* __restrict__ vrow = qkv + (size_t)(j + skey) * QKV_LD + 2 * D_IN + h * HDIM + sfs;
            __builtin_prefetch(vrow + 32 * QKV_LD, 0, 1);   // next block -> global_prefetch_b8
#pragma unroll
            for (int e = 0; e < 16; ++e)
                vshT[(sfs + e) * 32 + skey] = (_Float16)vrow[e];
        }
        __syncthreads();

        // ---- S = scale * Q K^T : two 16x16 tiles (b128 LDS + cvt) ---------
        auto loadB_K = [&](int t, int ks) -> v16h {
            const float* q = &kshF[(t * 16 + l16) * 64 + ks * 32 + kb];
            v16h b;
#pragma unroll
            for (int e = 0; e < 8; ++e) {
                b[e]     = (_Float16)q[e];
                b[e + 8] = (_Float16)q[16 + e];
            }
            return b;
        };
        v8f s0 = {}, s1 = {};
        s0 = __builtin_amdgcn_wmma_f32_16x16x32_f16(false, aq0, false, loadB_K(0,0), (short)0, s0, false, false);
        s0 = __builtin_amdgcn_wmma_f32_16x16x32_f16(false, aq1, false, loadB_K(0,1), (short)0, s0, false, false);
        s1 = __builtin_amdgcn_wmma_f32_16x16x32_f16(false, aq0, false, loadB_K(1,0), (short)0, s1, false, false);
        s1 = __builtin_amdgcn_wmma_f32_16x16x32_f16(false, aq1, false, loadB_K(1,1), (short)0, s1, false, false);

        // ---- streaming softmax (rows live per 16-lane half) ---------------
#pragma unroll
        for (int r = 0; r < 8; ++r) {
            float x0 = s0[r] * scale;
            float x1 = s1[r] * scale;
            float mx = fmaxf(x0, x1);
#pragma unroll
            for (int off = 1; off < 16; off <<= 1)
                mx = fmaxf(mx, __shfl_xor(mx, off, 32));
            const float mnew  = fmaxf(mrow[r], mx);
            const float alpha = __expf(mrow[r] - mnew);
            mrow[r] = mnew;
            const float p0 = __expf(x0 - mnew);
            const float p1 = __expf(x1 - mnew);
            s0[r] = p0; s1[r] = p1;
            float ls = p0 + p1;
#pragma unroll
            for (int off = 1; off < 16; off <<= 1)
                ls += __shfl_xor(ls, off, 32);
            lrow[r] = lrow[r] * alpha + ls;
            o0[r] *= alpha; o1[r] *= alpha; o2[r] *= alpha; o3[r] *= alpha;
        }

        // ---- P (C-layout) -> per-wave LDS -> A-fragment -------------------
#pragma unroll
        for (int r = 0; r < 8; ++r) {
            pSh[wid][(r + 8 * half) * 32 + l16]      = (_Float16)s0[r];
            pSh[wid][(r + 8 * half) * 32 + 16 + l16] = (_Float16)s1[r];
        }
        v16h ap;
        {
            const _Float16* q = &pSh[wid][l16 * 32 + kb];
            v8h lo = *(const v8h*)q;
            v8h hi = *(const v8h*)(q + 16);
            ap = __builtin_shufflevector(lo, hi, 0,1,2,3,4,5,6,7,8,9,10,11,12,13,14,15);
        }

        // ---- O += P @ V : four 16x16 feature tiles (b128 LDS) -------------
        auto loadB_V = [&](int fb) -> v16h {
            const _Float16* q = &vshT[(fb * 16 + l16) * 32 + kb];
            v8h lo = *(const v8h*)q;
            v8h hi = *(const v8h*)(q + 16);
            return __builtin_shufflevector(lo, hi, 0,1,2,3,4,5,6,7,8,9,10,11,12,13,14,15);
        };
        o0 = __builtin_amdgcn_wmma_f32_16x16x32_f16(false, ap, false, loadB_V(0), (short)0, o0, false, false);
        o1 = __builtin_amdgcn_wmma_f32_16x16x32_f16(false, ap, false, loadB_V(1), (short)0, o1, false, false);
        o2 = __builtin_amdgcn_wmma_f32_16x16x32_f16(false, ap, false, loadB_V(2), (short)0, o2, false, false);
        o3 = __builtin_amdgcn_wmma_f32_16x16x32_f16(false, ap, false, loadB_V(3), (short)0, o3, false, false);
        __syncthreads();
    }

    // ---- write partials (unnormalized O, per-row m/l) ---------------------
#pragma unroll
    for (int r = 0; r < 8; ++r) {
        const int row = qrow0 + r + 8 * half;
        float* __restrict__ op = opart + (((size_t)p * NHEAD + h) * N + row) * HDIM;
        op[ 0 + l16] = o0[r];
        op[16 + l16] = o1[r];
        op[32 + l16] = o2[r];
        op[48 + l16] = o3[r];
    }
    if (l16 == 0) {
#pragma unroll
        for (int r = 0; r < 8; ++r) {
            const int row = qrow0 + r + 8 * half;
            const size_t mi = ((size_t)p * NHEAD + h) * N + row;
            mpart[mi] = mrow[r];
            lpart[mi] = lrow[r];
        }
    }
}

// ---------------------------------------------------------------------------
// Combine key-partition partials: out = sum_p w_p O_p / sum_p w_p l_p,
// w_p = exp(m_p - max_p m_p).
// ---------------------------------------------------------------------------
__global__ void attn_combine(const float* __restrict__ opart,
                             const float* __restrict__ mpart,
                             const float* __restrict__ lpart,
                             float* __restrict__ attn, int N)
{
    int idx = blockIdx.x * blockDim.x + threadIdx.x;
    if (idx >= N * D_IN) return;
    const int row = idx / D_IN;
    const int c   = idx % D_IN;
    const int h   = c / HDIM;
    const int f   = c % HDIM;

    float mmax = -1e30f;
#pragma unroll
    for (int p = 0; p < PART; ++p)
        mmax = fmaxf(mmax, mpart[((size_t)p * NHEAD + h) * N + row]);
    float num = 0.0f, den = 0.0f;
#pragma unroll
    for (int p = 0; p < PART; ++p) {
        const size_t mi = ((size_t)p * NHEAD + h) * N + row;
        const float w = __expf(mpart[mi] - mmax);
        den += w * lpart[mi];
        num += w * opart[mi * HDIM + f];
    }
    attn[idx] = num / den;
}

// ---------------------------------------------------------------------------
// GCN helpers
// ---------------------------------------------------------------------------
__global__ void deg_init(float* __restrict__ deg, int N)
{
    int i = blockIdx.x * blockDim.x + threadIdx.x;
    if (i < N) deg[i] = 1.0f;               // self loop
}

__global__ void deg_count(const int* __restrict__ dst, float* __restrict__ deg, int E)
{
    int e = blockIdx.x * blockDim.x + threadIdx.x;
    if (e < E) atomicAdd(&deg[dst[e]], 1.0f);
}

__global__ void deg_rsqrt(float* __restrict__ deg, int N)
{
    int i = blockIdx.x * blockDim.x + threadIdx.x;
    if (i < N) deg[i] = rsqrtf(deg[i]);
}

__global__ void selfloop_init(const float* __restrict__ h,
                              const float* __restrict__ dinv,
                              float* __restrict__ out, int N, int F)
{
    int idx = blockIdx.x * blockDim.x + threadIdx.x;
    if (idx < N * F) {
        int n = idx / F;
        float dv = dinv[n];
        out[idx] = h[idx] * dv * dv;
    }
}

__global__ void edge_scatter(const float* __restrict__ h,
                             const float* __restrict__ dinv,
                             const int* __restrict__ src,
                             const int* __restrict__ dst,
                             float* __restrict__ out, int E, int F)
{
    int idx = blockIdx.x * blockDim.x + threadIdx.x;
    const int per = F >> 2;
    int e = idx / per;
    if (e >= E) return;
    int c = (idx % per) << 2;
    const int s = src[e];
    const int d = dst[e];
    const float nrm = dinv[s] * dinv[d];
    const float4 hv = *(const float4*)(h + (size_t)s * F + c);
    float* o = out + (size_t)d * F + c;
    atomicAdd(o + 0, hv.x * nrm);
    atomicAdd(o + 1, hv.y * nrm);
    atomicAdd(o + 2, hv.z * nrm);
    atomicAdd(o + 3, hv.w * nrm);
}

__global__ void bias_act(float* __restrict__ out, const float* __restrict__ b,
                         int N, int F, int doRelu)
{
    int idx = blockIdx.x * blockDim.x + threadIdx.x;
    if (idx < N * F) {
        float v = out[idx] + b[idx % F];
        out[idx] = doRelu ? fmaxf(v, 0.0f) : v;
    }
}

// ---------------------------------------------------------------------------
// Host-side launcher
// ---------------------------------------------------------------------------
extern "C" void kernel_launch(void* const* d_in, const int* in_sizes, int n_in,
                              void* d_out, int out_size, void* d_ws, size_t ws_size,
                              hipStream_t stream)
{
    const float* x     = (const float*)d_in[0];
    const int*   eidx  = (const int*)  d_in[1];
    const float* in_w  = (const float*)d_in[2];
    const float* in_b  = (const float*)d_in[3];
    const float* out_w = (const float*)d_in[4];
    const float* out_b = (const float*)d_in[5];
    const float* W1    = (const float*)d_in[6];
    const float* b1    = (const float*)d_in[7];
    const float* W2    = (const float*)d_in[8];
    const float* b2    = (const float*)d_in[9];

    const int N = in_sizes[0] / D_IN;       // 4096
    const int E = in_sizes[1] / 2;          // 131072
    const int* src = eidx;
    const int* dst = eidx + E;

    float* ws    = (float*)d_ws;
    float* qkv   = ws;                                   // N*768
    float* opart = qkv   + (size_t)N * QKV_LD;           // PART*NHEAD*N*HDIM
    float* mpart = opart + (size_t)PART * NHEAD * N * HDIM;  // PART*NHEAD*N
    float* lpart = mpart + (size_t)PART * NHEAD * N;     // PART*NHEAD*N
    float* attn  = lpart + (size_t)PART * NHEAD * N;     // N*256
    float* h0    = attn  + (size_t)N * D_IN;             // N*256
    float* hl    = h0    + (size_t)N * D_IN;             // N*256
    float* hagg  = hl    + (size_t)N * D_IN;             // N*256
    float* deg   = hagg  + (size_t)N * D_IN;             // N
    float* dout  = (float*)d_out;

    const int TPB = 256;

    // 1. QKV projection
    gemm_wmma_f16<<<(N / 16) * (QKV_LD / 64), 32, 0, stream>>>(x, in_w, in_b, qkv, N, D_IN, QKV_LD);

    // 2. Partitioned flash attention + combine
    attn_partial<<<dim3(N / 64, NHEAD, PART), 128, 0, stream>>>(qkv, opart, mpart, lpart, N);
    attn_combine<<<(N * D_IN + TPB - 1) / TPB, TPB, 0, stream>>>(opart, mpart, lpart, attn, N);

    // 3. Output projection
    gemm_wmma_f16<<<(N / 16) * (D_IN / 64), 32, 0, stream>>>(attn, out_w, out_b, h0, N, D_IN, D_IN);

    // 4. Degrees -> dinv
    deg_init <<<(N + TPB - 1) / TPB, TPB, 0, stream>>>(deg, N);
    deg_count<<<(E + TPB - 1) / TPB, TPB, 0, stream>>>(dst, deg, E);
    deg_rsqrt<<<(N + TPB - 1) / TPB, TPB, 0, stream>>>(deg, N);

    // 5. GCN layer 1
    gemm_wmma_f16<<<(N / 16) * (D_IN / 64), 32, 0, stream>>>(h0, W1, nullptr, hl, N, D_IN, D_IN);
    selfloop_init<<<(N * D_IN + TPB - 1) / TPB, TPB, 0, stream>>>(hl, deg, hagg, N, D_IN);
    edge_scatter<<<(E * (D_IN / 4) + TPB - 1) / TPB, TPB, 0, stream>>>(hl, deg, src, dst, hagg, E, D_IN);
    bias_act<<<(N * D_IN + TPB - 1) / TPB, TPB, 0, stream>>>(hagg, b1, N, D_IN, 1);

    // 6. GCN layer 2 -> d_out
    gemm_wmma_f16<<<(N / 16) * (D_IN / 64), 32, 0, stream>>>(hagg, W2, nullptr, hl, N, D_IN, D_IN);
    selfloop_init<<<(N * D_IN + TPB - 1) / TPB, TPB, 0, stream>>>(hl, deg, dout, N, D_IN);
    edge_scatter<<<(E * (D_IN / 4) + TPB - 1) / TPB, TPB, 0, stream>>>(hl, deg, src, dst, dout, E, D_IN);
    bias_act<<<(N * D_IN + TPB - 1) / TPB, TPB, 0, stream>>>(dout, b2, N, D_IN, 0);
}
